// Listener_34617436405769
// MI455X (gfx1250) — compile-verified, matrix operand order checked by reading, not run on hardware
//
#include <hip/hip_runtime.h>
#include <hip/hip_bf16.h>

#define DEV_INLINE __device__ __forceinline__

typedef __attribute__((ext_vector_type(16))) __bf16 v16bf;
typedef __attribute__((ext_vector_type(4)))  __bf16 v4bf;
typedef __attribute__((ext_vector_type(8)))  float  v8f;
typedef __attribute__((ext_vector_type(4)))  uint32_t u32x4;
typedef __attribute__((ext_vector_type(8)))  uint32_t u32x8;

namespace {
constexpr int kB   = 32;
constexpr int kT   = 1024;
constexpr int kF   = 80;
constexpr int kH   = 512;
constexpr int kT2  = 255;
constexpr int kF2  = 19;
constexpr int kD0  = 608;   // F2*32
constexpr int kD0p = 640;   // padded to multiple of 32
constexpr int kD   = 1024;  // 2*H
constexpr int kNG  = 2048;  // 4*H
constexpr int kM   = kB * kT2; // 8160
}

DEV_INLINE v8f wmma_bf16(v16bf a, v16bf b, v8f c) {
  // D(f32,16x16) = A(bf16,16x32) * B(bf16,32x16) + C
  return __builtin_amdgcn_wmma_f32_16x16x32_bf16(false, a, false, b, (short)0, c,
                                                 false, false);
}
DEV_INLINE float sigmoidf_(float x) { return 1.0f / (1.0f + __expf(-x)); }

// Async global->LDS copy of 8 bytes per lane (ASYNCcnt-tracked).
// ldsOff is the LDS byte address (low 32 bits of the generic shared pointer,
// per the flat->LDS aperture truncation rule, ISA 10.2).
DEV_INLINE void async_load_lds_b64(uint32_t ldsOff, const void* gaddr) {
  asm volatile("global_load_async_to_lds_b64 %0, %1, off"
               :: "v"(ldsOff), "v"((uint64_t)(uintptr_t)gaddr)
               : "memory");
}
DEV_INLINE void wait_asynccnt0() {
  asm volatile("s_wait_asynccnt 0x0" ::: "memory");
}

// Tensor Data Mover: DMA a 2D tile (rows x tile0 elements of 2B) from global
// into LDS. Descriptor per ISA 8.3/8.4: group0 = {flags, lds_addr, gaddr,
// type=2}; group1 = {data_size, tensor dims, tile dims, dim0 stride}.
// Issued by one wave; fence with s_wait_tensorcnt + workgroup barrier.
DEV_INLINE void tdm_load_tile_bf16(uint32_t ldsOff, const void* gaddr,
                                   uint32_t strideElems, uint32_t rows,
                                   uint32_t tile0, uint32_t tile1) {
  uint64_t ga = (uint64_t)(uintptr_t)gaddr;
  u32x4 g0;
  g0[0] = 1u;                                   // count=1, user descriptor
  g0[1] = ldsOff;                               // lds_addr (bytes)
  g0[2] = (uint32_t)ga;                         // global_addr[31:0]
  g0[3] = (uint32_t)(ga >> 32) | (2u << 30);    // global_addr[56:32] | type=2
  u32x8 g1;
  g1[0] = 1u << 16;                             // data_size=1 (2 bytes/elem)
  g1[1] = (strideElems & 0xFFFFu) << 16;        // tensor_dim0[15:0]
  g1[2] = (strideElems >> 16) | ((rows & 0xFFFFu) << 16); // dim0 hi | dim1 lo
  g1[3] = (rows >> 16) | (tile0 << 16);         // dim1 hi | tile_dim0
  g1[4] = tile1;                                // tile_dim1 (tile_dim2=0)
  g1[5] = strideElems;                          // tensor_dim0_stride[31:0]
  g1[6] = 0;                                    // stride hi | dim1_stride lo
  g1[7] = 0;
  asm volatile("tensor_load_to_lds %0, %1" :: "s"(g0), "s"(g1) : "memory");
}

// ---------------------------------------------------------------------------
// Zero a float buffer (state init each call; launch must be deterministic).
// ---------------------------------------------------------------------------
__global__ void zero_f32_kernel(float* p, int n) {
  int i = blockIdx.x * blockDim.x + threadIdx.x;
  if (i < n) p[i] = 0.0f;
}

// ---------------------------------------------------------------------------
// mask[b][t2] = any(audio[b, 4*t2 .. 4*t2+3, :] != 0)
// ---------------------------------------------------------------------------
__global__ void mask_kernel(const float* __restrict__ audio,
                            float* __restrict__ maskf,
                            float* __restrict__ mask_out) {
  int idx = blockIdx.x * blockDim.x + threadIdx.x;
  if (idx >= kM) return;
  int b = idx / kT2, t2 = idx % kT2;
  float any = 0.0f;
  for (int dt = 0; dt < 4; ++dt) {
    const float* row = audio + ((size_t)b * kT + 4 * t2 + dt) * kF;
    for (int f = 0; f < kF; ++f)
      if (row[f] != 0.0f) any = 1.0f;
  }
  maskf[idx] = any;
  mask_out[idx] = any;
}

// ---------------------------------------------------------------------------
// Fused conv1(3x3,s2) + conv2(3x3,s2). One block per (b, t2); x1 tile in LDS.
// Writes act0 as bf16 [M][640] (cols 608..639 zero-padded).
// ---------------------------------------------------------------------------
__global__ void __launch_bounds__(256)
conv_fused_kernel(const float* __restrict__ audio,
                  const float* __restrict__ k1, const float* __restrict__ b1,
                  const float* __restrict__ k2, const float* __restrict__ b2,
                  __bf16* __restrict__ act0) {
  __shared__ float x1t[3 * 39 * 32]; // 14976 B
  const int tid = threadIdx.x;
  const int b   = blockIdx.x / kT2;
  const int t2  = blockIdx.x % kT2;

  // Phase 1: x1 rows 2*t2 .. 2*t2+2 (each needs 3x3x1 conv over audio)
  for (int idx = tid; idx < 3 * 39 * 32; idx += 256) {
    int dt = idx / (39 * 32);
    int rem = idx % (39 * 32);
    int f1 = rem / 32, c = rem % 32;
    int t1 = 2 * t2 + dt;
    float acc = b1[c];
    #pragma unroll
    for (int kt = 0; kt < 3; ++kt)
      #pragma unroll
      for (int kf = 0; kf < 3; ++kf)
        acc += audio[((size_t)b * kT + 2 * t1 + kt) * kF + 2 * f1 + kf] *
               k1[(kt * 3 + kf) * 32 + c];
    x1t[idx] = acc;
  }
  __syncthreads();

  // Phase 2: conv2 outputs for this (b, t2): [19][32] -> act0 row
  for (int idx = tid; idx < kD0p; idx += 256) {
    float val = 0.0f;
    if (idx < kD0) {
      int f2 = idx / 32, c2 = idx % 32;
      float acc = b2[c2];
      for (int kt = 0; kt < 3; ++kt)
        for (int kf = 0; kf < 3; ++kf) {
          const float* xrow = &x1t[(kt * 39 + 2 * f2 + kf) * 32];
          const float* krow = &k2[((kt * 3 + kf) * 32) * 32 + c2];
          #pragma unroll 8
          for (int ci = 0; ci < 32; ++ci)
            acc += xrow[ci] * krow[ci * 32];
        }
      val = acc;
    }
    act0[(size_t)(b * kT2 + t2) * kD0p + idx] = (__bf16)val;
  }
}

// ---------------------------------------------------------------------------
// Convert f32 weight [K][N] -> transposed bf16 [N][Kp] (rows k>=K zero-padded)
// ---------------------------------------------------------------------------
__global__ void convertT_kernel(const float* __restrict__ src,
                                __bf16* __restrict__ dst,
                                int K, int N, int Kp) {
  int idx = blockIdx.x * blockDim.x + threadIdx.x;
  if (idx >= N * Kp) return;
  int n = idx / Kp, k = idx % Kp;
  float v = (k < K) ? src[(size_t)k * N + n] : 0.0f;
  dst[idx] = (__bf16)v;
}

// ---------------------------------------------------------------------------
// WMMA GEMM: out[M][N] = A[M][lda](bf16) * BT[N][ldb](bf16)^T + bias
// Block = 256 threads (8 waves). Block tile 32M x 256N; wave tile 16M x 64N.
// A K-chunk staged in LDS via global_load_async_to_lds (ASYNCcnt path).
// ---------------------------------------------------------------------------
__global__ void __launch_bounds__(256)
gemm_xz_kernel(const __bf16* __restrict__ A, const __bf16* __restrict__ BT,
               const float* __restrict__ bias, float* __restrict__ out,
               int N, int K, int lda, int ldb) {
  __shared__ __bf16 sA[32 * 32];
  const int tid  = threadIdx.x;
  const int lane = tid & 31, w = tid >> 5;
  const int msub = w & 1, nsub = w >> 1;
  const int m0 = blockIdx.x * 32;
  const int n0 = blockIdx.y * 256 + nsub * 64;

  v8f acc[4];
  #pragma unroll
  for (int j = 0; j < 4; ++j) acc[j] = {};

  const int lrow = (tid * 4) >> 5;   // cooperative A-tile load coords
  const int lcol = (tid * 4) & 31;
  const uint32_t ldsOff = (uint32_t)(uintptr_t)(&sA[tid * 4]);

  for (int k0 = 0; k0 < K; k0 += 32) {
    __syncthreads();
    async_load_lds_b64(ldsOff, A + (size_t)(m0 + lrow) * lda + k0 + lcol);
    if (k0 + 32 < K)
      __builtin_prefetch(A + (size_t)(m0 + lrow) * lda + k0 + 32 + lcol, 0, 1);
    wait_asynccnt0();
    __syncthreads();

    const int kh = (lane >> 4) << 4; // 0 or 16: K half per lane group
    v16bf bfv[4];
    #pragma unroll
    for (int j = 0; j < 4; ++j)
      bfv[j] =
          *(const v16bf*)(BT + (size_t)(n0 + j * 16 + (lane & 15)) * ldb + k0 + kh);
    v16bf af = *(const v16bf*)(&sA[(msub * 16 + (lane & 15)) * 32 + kh]);
    #pragma unroll
    for (int j = 0; j < 4; ++j) acc[j] = wmma_bf16(af, bfv[j], acc[j]);
  }

  // Epilogue: C layout — VGPR r: lanes0-15 M=r, lanes16-31 M=r+8; N=lane%16
  const int mrow = m0 + msub * 16 + ((lane >> 4) << 3);
  #pragma unroll
  for (int j = 0; j < 4; ++j) {
    int col = n0 + j * 16 + (lane & 15);
    float bc = bias[col];
    #pragma unroll
    for (int r = 0; r < 8; ++r)
      out[(size_t)(mrow + r) * N + col] = acc[j][r] + bc;
  }
}

// ---------------------------------------------------------------------------
// WMMA GEMM + bias + batchnorm + relu. A K-chunk staged in LDS via the
// Tensor Data Mover (TENSORcnt path), issued by wave 0 only. Writes
// next-layer activation (bf16) and optionally f32 to d_out (final layer).
// ---------------------------------------------------------------------------
__global__ void __launch_bounds__(256)
gemm_proj_kernel(const __bf16* __restrict__ A, const __bf16* __restrict__ BT,
                 const float* __restrict__ bias,
                 const float* __restrict__ bn_g, const float* __restrict__ bn_beta,
                 const float* __restrict__ bn_mu, const float* __restrict__ bn_var,
                 __bf16* __restrict__ actOut, float* __restrict__ outF,
                 int N, int K, int lda, int ldb) {
  __shared__ __bf16 sA[32 * 32];
  const int tid  = threadIdx.x;
  const int lane = tid & 31, w = tid >> 5;
  const int msub = w & 1, nsub = w >> 1;
  const int m0 = blockIdx.x * 32;
  const int n0 = blockIdx.y * 256 + nsub * 64;

  v8f acc[4];
  #pragma unroll
  for (int j = 0; j < 4; ++j) acc[j] = {};

  const uint32_t ldsBase = (uint32_t)(uintptr_t)(&sA[0]);

  for (int k0 = 0; k0 < K; k0 += 32) {
    __syncthreads();
    if (tid < 32) { // one wave drives the TDM; EXEC is ignored by tensor ops
      tdm_load_tile_bf16(ldsBase, A + (size_t)m0 * lda + k0,
                         (uint32_t)lda, 32u, 32u, 32u);
      __builtin_amdgcn_s_wait_tensorcnt(0);
    }
    __syncthreads();

    const int kh = (lane >> 4) << 4;
    v16bf bfv[4];
    #pragma unroll
    for (int j = 0; j < 4; ++j)
      bfv[j] =
          *(const v16bf*)(BT + (size_t)(n0 + j * 16 + (lane & 15)) * ldb + k0 + kh);
    v16bf af = *(const v16bf*)(&sA[(msub * 16 + (lane & 15)) * 32 + kh]);
    #pragma unroll
    for (int j = 0; j < 4; ++j) acc[j] = wmma_bf16(af, bfv[j], acc[j]);
  }

  const int mrow = m0 + msub * 16 + ((lane >> 4) << 3);
  #pragma unroll
  for (int j = 0; j < 4; ++j) {
    int col = n0 + j * 16 + (lane & 15);
    float bc = bias[col];
    float sc = bn_g[col] * rsqrtf(bn_var[col] + 1e-3f);
    float mu = bn_mu[col], be = bn_beta[col];
    #pragma unroll
    for (int r = 0; r < 8; ++r) {
      float v = acc[j][r] + bc;
      v = fmaf(sc, v - mu, be);
      v = fmaxf(v, 0.0f);
      actOut[(size_t)(mrow + r) * N + col] = (__bf16)v;
      if (outF) outF[(size_t)(mrow + r) * N + col] = v;
    }
  }
}

// ---------------------------------------------------------------------------
// Masked LSTM scan (one direction). Single workgroup, 32 waves (1024 thr).
// Wave w owns hidden units j in [16w, 16w+16): computes the 4 gate columns
// for those units (no cross-wave gate exchange). h (bf16) lives in LDS and is
// the shared WMMA A operand; c and h carried in VGPRs across all 255 steps.
// U pre-transposed to [2048][512] bf16 -> contiguous per-lane B fragments.
// ---------------------------------------------------------------------------
__global__ void __launch_bounds__(1024)
lstm_scan_kernel(const float* __restrict__ xz,    // [M][2048], m = b*T2 + t
                 const __bf16* __restrict__ UT,   // [2048][512]
                 const float* __restrict__ maskf, // [B][T2]
                 float* __restrict__ stH, float* __restrict__ stC, // [B][512]
                 __bf16* __restrict__ ysc,        // [M][1024]
                 int dir) {
  __shared__ __bf16 shH[kB * kH]; // 32 KB
  __shared__ float  shM[kB];
  const int tid  = threadIdx.x;
  const int lane = tid & 31, w = tid >> 5;
  const int j    = (w << 4) + (lane & 15); // hidden unit for this lane

  float cst[16], hpr[16];
  #pragma unroll
  for (int p = 0; p < 16; ++p) {
    int b = ((p >> 3) << 4) + ((lane >> 4) << 3) + (p & 7);
    hpr[p] = stH[b * kH + j];
    cst[p] = stC[b * kH + j];
  }
  #pragma unroll
  for (int q = 0; q < 16; ++q) {
    int idx = tid * 16 + q; // covers 32*512
    shH[idx] = (__bf16)stH[idx];
  }
  __syncthreads();

  #pragma unroll 1
  for (int s = 0; s < kT2; ++s) {
    const int t_in = dir ? (kT2 - 1 - s) : s;
    if (tid < kB) shM[tid] = maskf[tid * kT2 + t_in];
    __syncthreads(); // shM ready + previous step's shH writes visible

    v8f acc[2][4];
    #pragma unroll
    for (int mt = 0; mt < 2; ++mt)
      #pragma unroll
      for (int g = 0; g < 4; ++g) acc[mt][g] = {};

    #pragma unroll 2
    for (int kc = 0; kc < 16; ++kc) {
      const int kh = (kc << 5) + ((lane >> 4) << 4);
      v16bf a0 = *(const v16bf*)(&shH[(lane & 15) * kH + kh]);
      v16bf a1 = *(const v16bf*)(&shH[(16 + (lane & 15)) * kH + kh]);
      v16bf bfv[4];
      #pragma unroll
      for (int g = 0; g < 4; ++g)
        bfv[g] = *(const v16bf*)(
            UT + (size_t)((g << 9) + (w << 4) + (lane & 15)) * kH + kh);
      #pragma unroll
      for (int g = 0; g < 4; ++g) {
        acc[0][g] = wmma_bf16(a0, bfv[g], acc[0][g]);
        acc[1][g] = wmma_bf16(a1, bfv[g], acc[1][g]);
      }
    }

    // Gate nonlinearity + masked state update for 16 (batch) entries per lane.
    #pragma unroll
    for (int p = 0; p < 16; ++p) {
      const int mt = p >> 3, r = p & 7;
      const int b = (mt << 4) + ((lane >> 4) << 3) + r;
      const float* zrow = xz + ((size_t)b * kT2 + t_in) * kNG;
      float zi = acc[mt][0][r] + zrow[j];
      float zf = acc[mt][1][r] + zrow[kH + j];
      float zg = acc[mt][2][r] + zrow[2 * kH + j];
      float zo = acc[mt][3][r] + zrow[3 * kH + j];
      float cn = sigmoidf_(zf) * cst[p] + sigmoidf_(zi) * tanhf(zg);
      float hn = sigmoidf_(zo) * tanhf(cn);
      bool mk = shM[b] > 0.5f;
      hpr[p] = mk ? hn : hpr[p];
      cst[p] = mk ? cn : cst[p];
      ysc[((size_t)b * kT2 + s) * kD + (dir << 9) + j] = (__bf16)hpr[p];
    }
    __syncthreads(); // all waves done reading shH for this step
    #pragma unroll
    for (int p = 0; p < 16; ++p) {
      int b = ((p >> 3) << 4) + ((lane >> 4) << 3) + (p & 7);
      shH[b * kH + j] = (__bf16)hpr[p];
    }
  }

  #pragma unroll
  for (int p = 0; p < 16; ++p) {
    int b = ((p >> 3) << 4) + ((lane >> 4) << 3) + (p & 7);
    stH[b * kH + j] = hpr[p];
    stC[b * kH + j] = cst[p];
  }
}

// ---------------------------------------------------------------------------
// Concatenate final (fh,bh) / (fc,bc) into d_out state regions.
// ---------------------------------------------------------------------------
__global__ void finalize_state_kernel(const float* __restrict__ fh,
                                      const float* __restrict__ fc,
                                      const float* __restrict__ bh,
                                      const float* __restrict__ bc,
                                      float* __restrict__ outH,
                                      float* __restrict__ outC) {
  int idx = blockIdx.x * blockDim.x + threadIdx.x;
  if (idx >= kB * kD) return;
  int b = idx >> 10, k = idx & 1023;
  outH[idx] = (k < kH) ? fh[b * kH + k] : bh[b * kH + (k - kH)];
  outC[idx] = (k < kH) ? fc[b * kH + k] : bc[b * kH + (k - kH)];
}

// ---------------------------------------------------------------------------
extern "C" void kernel_launch(void* const* d_in, const int* in_sizes, int n_in,
                              void* d_out, int out_size, void* d_ws, size_t ws_size,
                              hipStream_t stream) {
  (void)in_sizes; (void)n_in; (void)out_size; (void)ws_size;

  const float* audio = (const float*)d_in[0];
  const float* c1k = (const float*)d_in[1];
  const float* c1b = (const float*)d_in[2];
  const float* c2k = (const float*)d_in[3];
  const float* c2b = (const float*)d_in[4];

  // ---- workspace layout --------------------------------------------------
  char* ws = (char*)d_ws;
  size_t off = 0;
  auto alloc = [&](size_t bytes) {
    size_t r = off;
    off += (bytes + 255) & ~(size_t)255;
    return r;
  };
  size_t oMask = alloc((size_t)kM * 4);
  size_t oAct0 = alloc((size_t)kM * kD0p * 2);
  size_t oActA = alloc((size_t)kM * kD * 2);
  size_t oXz   = alloc((size_t)kM * kNG * 4);
  size_t oYsc  = alloc((size_t)kM * kD * 2);
  size_t oSt   = alloc((size_t)4 * kB * kH * 4);
  size_t oWT[3][2], oUT[3][2], oPwT[3];
  for (int i = 0; i < 3; ++i) {
    for (int d = 0; d < 2; ++d) {
      oWT[i][d] = alloc((size_t)kNG * kD * 2);
      oUT[i][d] = alloc((size_t)kNG * kH * 2);
    }
    oPwT[i] = alloc((size_t)kD * kD * 2);
  }

  float*  maskf = (float*)(ws + oMask);
  __bf16* act0  = (__bf16*)(ws + oAct0);
  __bf16* actA  = (__bf16*)(ws + oActA);
  float*  xz    = (float*)(ws + oXz);
  __bf16* ysc   = (__bf16*)(ws + oYsc);
  float*  stFh  = (float*)(ws + oSt);
  float*  stFc  = stFh + kB * kH;
  float*  stBh  = stFc + kB * kH;
  float*  stBc  = stBh + kB * kH;

  float* outX    = (float*)d_out;                  // [M][1024]
  float* outMask = outX + (size_t)kM * kD;         // [M]
  float* outH    = outMask + kM;                   // [B][1024]
  float* outC    = outH + (size_t)kB * kD;         // [B][1024]

  // ---- front end ---------------------------------------------------------
  zero_f32_kernel<<<(4 * kB * kH + 255) / 256, 256, 0, stream>>>(stFh, 4 * kB * kH);
  mask_kernel<<<(kM + 255) / 256, 256, 0, stream>>>(audio, maskf, outMask);
  conv_fused_kernel<<<kM, 256, 0, stream>>>(audio, c1k, c1b, c2k, c2b, act0);

  // ---- weight conversion (f32 -> transposed bf16) ------------------------
  for (int i = 0; i < 3; ++i) {
    const int din = (i == 0) ? kD0 : kD;
    const int Kp  = (i == 0) ? kD0p : kD;
    const float* Wf = (const float*)d_in[5 + i * 12 + 0];
    const float* Uf = (const float*)d_in[5 + i * 12 + 1];
    const float* Wb = (const float*)d_in[5 + i * 12 + 3];
    const float* Ub = (const float*)d_in[5 + i * 12 + 4];
    const float* Pw = (const float*)d_in[5 + i * 12 + 6];
    int nW = kNG * Kp, nU = kNG * kH, nP = kD * kD;
    convertT_kernel<<<(nW + 255) / 256, 256, 0, stream>>>(Wf, (__bf16*)(ws + oWT[i][0]), din, kNG, Kp);
    convertT_kernel<<<(nW + 255) / 256, 256, 0, stream>>>(Wb, (__bf16*)(ws + oWT[i][1]), din, kNG, Kp);
    convertT_kernel<<<(nU + 255) / 256, 256, 0, stream>>>(Uf, (__bf16*)(ws + oUT[i][0]), kH, kNG, kH);
    convertT_kernel<<<(nU + 255) / 256, 256, 0, stream>>>(Ub, (__bf16*)(ws + oUT[i][1]), kH, kNG, kH);
    convertT_kernel<<<(nP + 255) / 256, 256, 0, stream>>>(Pw, (__bf16*)(ws + oPwT[i]), kD, kD, kD);
  }

  // ---- layers ------------------------------------------------------------
  for (int i = 0; i < 3; ++i) {
    const int Kp = (i == 0) ? kD0p : kD;
    const __bf16* act = (i == 0) ? act0 : actA;
    const float* bfv  = (const float*)d_in[5 + i * 12 + 2];
    const float* bbv  = (const float*)d_in[5 + i * 12 + 5];
    const float* Pb   = (const float*)d_in[5 + i * 12 + 7];
    const float* bnG  = (const float*)d_in[5 + i * 12 + 8];
    const float* bnB  = (const float*)d_in[5 + i * 12 + 9];
    const float* bnM  = (const float*)d_in[5 + i * 12 + 10];
    const float* bnV  = (const float*)d_in[5 + i * 12 + 11];

    for (int dir = 0; dir < 2; ++dir) {
      const __bf16* WT = (const __bf16*)(ws + oWT[i][dir]);
      const __bf16* UT = (const __bf16*)(ws + oUT[i][dir]);
      const float* bias = dir ? bbv : bfv;
      float* stH = dir ? stBh : stFh;
      float* stC = dir ? stBc : stFc;
      // xz = act @ W + b  : [8160 x Kp] x [Kp x 2048]
      gemm_xz_kernel<<<dim3(kM / 32, kNG / 256), 256, 0, stream>>>(
          act, WT, bias, xz, kNG, Kp, Kp, Kp);
      // sequential masked scan (WMMA recurrence)
      lstm_scan_kernel<<<1, 1024, 0, stream>>>(xz, UT, maskf, stH, stC, ysc, dir);
    }

    // projection + BN + relu : [8160 x 1024] x [1024 x 1024]
    gemm_proj_kernel<<<dim3(kM / 32, kD / 256), 256, 0, stream>>>(
        ysc, (const __bf16*)(ws + oPwT[i]), Pb, bnG, bnB, bnM, bnV,
        actA, (i == 2) ? outX : nullptr, kD, kD, kD, kD);
  }

  finalize_state_kernel<<<(kB * kD + 255) / 256, 256, 0, stream>>>(
      stFh, stFc, stBh, stBc, outH, outC);
}
